// Spiking_SwinTransformerBlock3D_44650480009379
// MI455X (gfx1250) — matrix operand order repaired and checked by reading, not compiled, not run on hardware
//
#include <hip/hip_runtime.h>

// ---------------------------------------------------------------------------
// Spiking Swin 3D window-attention block for MI455X (gfx1250), wave32 + WMMA.
//
// Roofline: ~17 GFLOP vs ~1.1 GB mandatory traffic at 23.3 TB/s -> memory
// bound, dominated by the 268 MB attn write. attn is produced, written, and
// consumed (attn@v) in one kernel. This revision adds WGP-level k/v reuse:
// one 16-wave workgroup per (batch, head) stages the 16 KB k and v panels
// into LDS once via GLOBAL_LOAD_ASYNC_TO_LDS_B128 (ASYNCcnt), removing the
// 16x panel re-read of the previous version.
// ---------------------------------------------------------------------------

typedef __attribute__((ext_vector_type(16))) _Float16 v16h;
typedef __attribute__((ext_vector_type(8)))  float    v8f;

#define T_DIM   4
#define C_DIM   128
#define NPOS    16384            // B*H*W spatial positions
#define QKV_ELEMS 33554432L      // 4 * (4*256*8*8*128)
#define OUT_ELEMS 8388608L       // 256*256*128
#define SCALE_QK 0.17677669529663687f   // 32^-0.5

// Async 16-byte global->LDS copy (gfx1250, tracked by ASYNCcnt).
__device__ __forceinline__ void async_copy_b128(unsigned lds_off, const void* gaddr) {
    asm volatile("global_load_async_to_lds_b128 %0, %1, off"
                 :: "v"(lds_off), "v"(gaddr)
                 : "memory");
}
__device__ __forceinline__ void wait_asynccnt0() {
    asm volatile("s_wait_asynccnt 0" ::: "memory");
}

// ---------------------------------------------------------------------------
// Kernel A: QKV projection + LIF, fused.
// grid = 1024 blocks (16 spatial positions each), 128 threads (4 waves).
// LIF recurrence over T runs directly in WMMA accumulator registers; binary
// f16 spikes are scattered into the literal-reshape attention layout.
// ---------------------------------------------------------------------------
__global__ __launch_bounds__(128)
void qkv_lif_kernel(const float* __restrict__ x,
                    const float* __restrict__ Wq,
                    const float* __restrict__ Wk,
                    const float* __restrict__ Wv,
                    _Float16* __restrict__ qh,
                    _Float16* __restrict__ kh,
                    _Float16* __restrict__ vh)
{
    const int p0   = blockIdx.x * 16;
    const int wave = threadIdx.x >> 5;
    const int lane = threadIdx.x & 31;
    const int r    = lane & 15;       // A row / B col / D col within tile
    const int hi   = lane >> 4;       // lane-group select

    for (int pi = wave * 6; pi < wave * 6 + 6; ++pi) {
        const int wsel = pi >> 3;             // 0=q 1=k 2=v
        const int n0   = (pi & 7) * 16;       // output-channel tile
        const float* W = (wsel == 0) ? Wq : ((wsel == 1) ? Wk : Wv);
        _Float16* dst  = (wsel == 0) ? qh : ((wsel == 1) ? kh : vh);

        v8f vstate = {};                      // LIF membrane, per (p,d) element
        for (int t = 0; t < T_DIM; ++t) {
            v8f acc = {};
            for (int k0 = 0; k0 < C_DIM; k0 += 32) {
                // A fragment: x rows (t*NPOS+p0 .. +15), K slice of 32, f32->f16
                v16h a;
                {
                    const float* pa = x + (long)(t * NPOS + p0 + r) * C_DIM + k0 + hi * 16;
                    #pragma unroll
                    for (int i = 0; i < 8; ++i) a[i] = (_Float16)pa[i];
                    #pragma unroll
                    for (int i = 0; i < 8; ++i) a[8 + i] = (_Float16)pa[8 + i];
                }
                // B fragment: B[k][n] = W[n0+n][k0+k] (out = x @ W^T)
                v16h b;
                {
                    const float* pb = W + (long)(n0 + r) * C_DIM + k0 + hi * 16;
                    #pragma unroll
                    for (int i = 0; i < 16; ++i) b[i] = (_Float16)pb[i];
                }
                acc = __builtin_amdgcn_wmma_f32_16x16x32_f16(
                        false, a, false, b, (short)0, acc, false, false);
            }
            // LIF step (tau=2, vth=1, hard reset) + scatter spike stores
            #pragma unroll
            for (int j = 0; j < 8; ++j) {
                float vv = 0.5f * (vstate[j] + acc[j]);
                float s  = (vv >= 1.0f) ? 1.0f : 0.0f;
                vstate[j] = vv * (1.0f - s);
                const int p  = p0 + j + hi * 8;        // spatial position
                const int d  = n0 + r;                 // channel
                const int b2 = t * 64 + (p >> 8);      // literal-reshape batch
                const int h2 = (p >> 6) & 3;           // head  (= b & 3)
                const int n2 = ((p >> 3) & 7) * 32 + (p & 7) * 4 + (d >> 5);
                const int d2 = d & 31;
                dst[(((long)(b2 * 4 + h2) * 256 + n2) << 5) + d2] = (_Float16)s;
            }
        }
    }
}

// ---------------------------------------------------------------------------
// Kernel B: fused attention. One 16-wave workgroup per (b2, head).
// k and v panels (16 KB each) are async-staged into LDS once and shared by
// all 16 row-tile waves. attn = scale*(q@k^T)+rpb is written once to d_out
// and immediately consumed (via wave-private f16 LDS restage) for attn@v.
// ---------------------------------------------------------------------------
__global__ __launch_bounds__(512)
void attn_fused_kernel(const _Float16* __restrict__ qh,
                       const _Float16* __restrict__ kh,
                       const _Float16* __restrict__ vh,
                       const float* __restrict__ rpb_table,
                       float* __restrict__ attn_out,   // (256,4,256,256) f32
                       _Float16* __restrict__ y)       // (T,B,H,W,C) f16
{
    __shared__ _Float16 Ksm[256 * 32];          // 16 KB k panel
    __shared__ _Float16 Vsm[256 * 32];          // 16 KB v panel
    __shared__ _Float16 Astage[16][16 * 32];    // per-wave attn restage (16 KB)

    const int bh   = blockIdx.x;                // b2*4 + head
    const int hd2  = bh & 3;
    const int wave = threadIdx.x >> 5;
    const int lane = threadIdx.x & 31;
    const int r    = lane & 15;
    const int hi   = lane >> 4;
    const int m0   = wave * 16;                 // this wave's attn row tile

    const long base = (long)bh * 256 * 32;
    const _Float16* qb = qh + base;
    float* attnb = attn_out + (long)bh * 256 * 256;

    // ---- Async stage k and v panels into LDS (2 x 16B per thread each) ----
    {
        const int tid = threadIdx.x;
        #pragma unroll
        for (int it = 0; it < 2; ++it) {
            const int off = (tid + it * 512) * 8;        // halves (16 B)
            async_copy_b128((unsigned)(uintptr_t)&Ksm[off], kh + base + off);
            async_copy_b128((unsigned)(uintptr_t)&Vsm[off], vh + base + off);
        }
        wait_asynccnt0();
    }
    __syncthreads();

    // Loop-invariant A fragment: q rows m0..m0+15, K = head_dim (32).
    v16h qa;
    {
        const _Float16* pa = qb + (m0 + r) * 32 + hi * 16;
        #pragma unroll
        for (int i = 0; i < 8; ++i) qa[i] = pa[i];
        #pragma unroll
        for (int i = 0; i < 8; ++i) qa[8 + i] = pa[8 + i];
    }

    _Float16* as = &Astage[wave][0];
    v8f oacc0 = {}, oacc1 = {};                 // out tile 16x32 (two N halves)

    for (int c0 = 0; c0 < 256; c0 += 32) {      // attn col-tile pairs (= K of GEMM2)
        #pragma unroll
        for (int half = 0; half < 2; ++half) {
            const int cc0 = c0 + half * 16;
            // B fragment from LDS k: B[k][n] = K[(cc0+n)*32 + k] (contiguous in k)
            v16h bk;
            {
                const _Float16* pb = &Ksm[(cc0 + r) * 32 + hi * 16];
                #pragma unroll
                for (int i = 0; i < 16; ++i) bk[i] = pb[i];
            }
            v8f acc = {};
            acc = __builtin_amdgcn_wmma_f32_16x16x32_f16(
                    false, qa, false, bk, (short)0, acc, false, false);

            // Epilogue: scale, relative-position bias, store attn, restage f16.
            const int m  = cc0 + r;                  // attn column
            const int md = m >> 6, mh = (m >> 3) & 7, mw = m & 7;
            #pragma unroll
            for (int j = 0; j < 8; ++j) {
                const int n  = m0 + j + hi * 8;      // attn row
                const int nd = n >> 6, nh = (n >> 3) & 7, nw = n & 7;
                const int rpi = (nd - md + 3) * 225 + (nh - mh + 7) * 15 + (nw - mw + 7);
                float av = acc[j] * SCALE_QK + rpb_table[rpi * 4 + hd2];
                attnb[(long)n * 256 + m] = av;
                as[(j + hi * 8) * 32 + half * 16 + r] = (_Float16)av;
            }
        }
        // Restaged attn tile (16x32) as A fragment (wave-private, no barrier).
        v16h aa;
        {
            const _Float16* pa = as + r * 32 + hi * 16;
            #pragma unroll
            for (int i = 0; i < 8; ++i) aa[i] = pa[i];
            #pragma unroll
            for (int i = 0; i < 8; ++i) aa[8 + i] = pa[8 + i];
        }
        // B fragments from LDS v: B[k][n] = V[(c0+k)*32 + n]
        #pragma unroll
        for (int nt = 0; nt < 2; ++nt) {
            v16h bv;
            #pragma unroll
            for (int i = 0; i < 16; ++i)
                bv[i] = Vsm[(c0 + hi * 16 + i) * 32 + nt * 16 + r];
            if (nt == 0)
                oacc0 = __builtin_amdgcn_wmma_f32_16x16x32_f16(
                          false, aa, false, bv, (short)0, oacc0, false, false);
            else
                oacc1 = __builtin_amdgcn_wmma_f32_16x16x32_f16(
                          false, aa, false, bv, (short)0, oacc1, false, false);
        }
    }

    // Scatter out rows back to (T,B,H,W,C) layout for the projection stage.
    const int b2 = bh >> 2;
    #pragma unroll
    for (int j = 0; j < 8; ++j) {
        const int n  = m0 + j + hi * 8;
        const int t3 = n >> 6, h3 = (n >> 3) & 7, w3 = n & 7;
        const long rowbase =
            ((((long)t3 * 256 + b2) * 8 + h3) * 8 + w3) * 128 + (long)hd2 * 32;
        y[rowbase + r]      = (_Float16)oacc0[j];
        y[rowbase + 16 + r] = (_Float16)oacc1[j];
    }
}

// ---------------------------------------------------------------------------
// Kernel C: output projection + bias + LIF, fused (same tiling as kernel A).
// ---------------------------------------------------------------------------
__global__ __launch_bounds__(128)
void proj_lif_kernel(const _Float16* __restrict__ y,
                     const float* __restrict__ Wproj,
                     const float* __restrict__ bproj,
                     float* __restrict__ out)
{
    const int p0   = blockIdx.x * 16;
    const int wave = threadIdx.x >> 5;
    const int lane = threadIdx.x & 31;
    const int r    = lane & 15;
    const int hi   = lane >> 4;

    for (int nt = wave * 2; nt < wave * 2 + 2; ++nt) {
        const int n0 = nt * 16;
        const float bias = bproj[n0 + r];     // D-tile column = lane&15
        v8f vstate = {};
        for (int t = 0; t < T_DIM; ++t) {
            v8f acc = {};
            for (int k0 = 0; k0 < C_DIM; k0 += 32) {
                v16h a;
                {
                    const _Float16* pa = y + (long)(t * NPOS + p0 + r) * C_DIM + k0 + hi * 16;
                    #pragma unroll
                    for (int i = 0; i < 8; ++i) a[i] = pa[i];
                    #pragma unroll
                    for (int i = 0; i < 8; ++i) a[8 + i] = pa[8 + i];
                }
                v16h b;
                {
                    const float* pb = Wproj + (long)(n0 + r) * C_DIM + k0 + hi * 16;
                    #pragma unroll
                    for (int i = 0; i < 16; ++i) b[i] = (_Float16)pb[i];
                }
                acc = __builtin_amdgcn_wmma_f32_16x16x32_f16(
                        false, a, false, b, (short)0, acc, false, false);
            }
            #pragma unroll
            for (int j = 0; j < 8; ++j) {
                float val = acc[j] + bias;
                float vv  = 0.5f * (vstate[j] + val);
                float s   = (vv >= 1.0f) ? 1.0f : 0.0f;
                vstate[j] = vv * (1.0f - s);
                const int p = p0 + j + hi * 8;
                out[(long)(t * NPOS + p) * C_DIM + n0 + r] = s;
            }
        }
    }
}

// ---------------------------------------------------------------------------
// Launcher
// Inputs: x, Wq, Wk, Wv, rpb_table, Wproj, bproj (all f32).
// Output: out (8,388,608 f32) ++ attn (67,108,864 f32).
// Workspace: q/k/v spikes + proj input, all f16 (268 MB total).
// ---------------------------------------------------------------------------
extern "C" void kernel_launch(void* const* d_in, const int* in_sizes, int n_in,
                              void* d_out, int out_size, void* d_ws, size_t ws_size,
                              hipStream_t stream) {
    const float* x     = (const float*)d_in[0];
    const float* Wq    = (const float*)d_in[1];
    const float* Wk    = (const float*)d_in[2];
    const float* Wv    = (const float*)d_in[3];
    const float* rpbt  = (const float*)d_in[4];
    const float* Wproj = (const float*)d_in[5];
    const float* bproj = (const float*)d_in[6];

    float* out_spk  = (float*)d_out;
    float* attn_out = out_spk + OUT_ELEMS;

    _Float16* qh = (_Float16*)d_ws;
    _Float16* kh = qh + QKV_ELEMS;
    _Float16* vh = kh + QKV_ELEMS;
    _Float16* y  = vh + QKV_ELEMS;

    qkv_lif_kernel<<<NPOS / 16, 128, 0, stream>>>(x, Wq, Wk, Wv, qh, kh, vh);
    attn_fused_kernel<<<1024, 512, 0, stream>>>(qh, kh, vh, rpbt, attn_out, y);
    proj_lif_kernel<<<NPOS / 16, 128, 0, stream>>>(y, Wproj, bproj, out_spk);
}
